// RationalQuadraticSplineCoupling_13915694039132
// MI455X (gfx1250) — matrix operand order repaired and checked by reading, not compiled, not run on hardware
//
#include <hip/hip_runtime.h>
#include <hip/hip_bf16.h>
#include <math.h>

// Problem constants (from reference)
#define DIM        128
#define HALFD      64
#define HIDDEN     256
#define NBINS      8
#define NSTAT      23            // 3*NBINS-1
#define OUTW       (NSTAT*HALFD) // 1472
#define ROWS_TOTAL 32768         // 4*1024*8
#define RPB        32            // rows per block

typedef __attribute__((ext_vector_type(16))) _Float16 v16h;
typedef __attribute__((ext_vector_type(8)))  _Float16 v8h;
typedef __attribute__((ext_vector_type(8)))  float    v8f;

// ---- WMMA fragment loaders (layouts per CDNA5 ISA 7.12.2, wave32) ----

// 16-bit A 16x32 (MxK): lanes 0-15: m=lane, V0..3=K0..7, V4..7=K16..23;
//                       lanes 16-31: m=lane-16, V0..3=K8..15, V4..7=K24..31.
__device__ inline v16h load_a_f16(const _Float16* __restrict__ base, int ld,
                                  int row0, int kbase, int lane) {
    const int h = lane >> 4, m = lane & 15;
    const _Float16* p = base + (size_t)(row0 + m) * ld + kbase + h * 8;
    v8h lo = *(const v8h*)p;          // K = kbase + h*8 + 0..7
    v8h hi = *(const v8h*)(p + 16);   // K = kbase + 16 + h*8 + 0..7
    v16h a;
#pragma unroll
    for (int i = 0; i < 8; ++i) { a[i] = lo[i]; a[i + 8] = hi[i]; }
    return a;
}

// Same A layout but converting from f32 source (x0) on the fly.
__device__ inline v16h load_a_f32cvt(const float* __restrict__ base, int ld,
                                     int row0, int kbase, int lane) {
    const int h = lane >> 4, m = lane & 15;
    const float* p = base + (size_t)(row0 + m) * ld + kbase + h * 8;
    v8f lo = *(const v8f*)p;
    v8f hi = *(const v8f*)(p + 16);
    v16h a;
#pragma unroll
    for (int i = 0; i < 8; ++i) { a[i] = (_Float16)lo[i]; a[i + 8] = (_Float16)hi[i]; }
    return a;
}

// 16-bit B 32x16 (KxN): lane n=lane&15; elems 0..7 hold K=kbase+h*8+0..7,
// elems 8..15 hold K=kbase+16+h*8+0..7 (same K order as A).
// Source memory holds W[col][k] row-major (w^T of the GEMM's B).
__device__ inline v16h load_b_f16(const _Float16* __restrict__ base, int ld,
                                  int col0, int kbase, int lane) {
    const int h = lane >> 4, n = lane & 15;
    const _Float16* p = base + (size_t)(col0 + n) * ld + kbase + h * 8;
    v8h lo = *(const v8h*)p;
    v8h hi = *(const v8h*)(p + 16);
    v16h b;
#pragma unroll
    for (int i = 0; i < 8; ++i) { b[i] = lo[i]; b[i + 8] = hi[i]; }
    return b;
}

// ---- kernel 0: weight f32 -> f16 conversion (one-time, tiny) ----
__global__ void k_cvt(const float* __restrict__ w1, const float* __restrict__ w2,
                      _Float16* __restrict__ w1h, _Float16* __restrict__ w2h) {
    int i = blockIdx.x * blockDim.x + threadIdx.x;
    if (i < HIDDEN * HALFD)  w1h[i] = (_Float16)w1[i];
    if (i < OUTW * HIDDEN)   w2h[i] = (_Float16)w2[i];
}

// ---- kernel 1: x0 passthrough + GEMM1 (WMMA) + LayerNorm + ReLU -> hb (f16) ----
__global__ __launch_bounds__(256) void k_gemm1_ln(
        const float* __restrict__ x, const _Float16* __restrict__ w1h,
        const float* __restrict__ ln_scale, const float* __restrict__ ln_bias,
        _Float16* __restrict__ hb, float* __restrict__ out) {
    __shared__ float hbuf[RPB * HIDDEN]; // 32 KB
    const int tid = threadIdx.x, lane = tid & 31, wave = tid >> 5;
    const int rowBase = blockIdx.x * RPB;

    // copy x0 -> out[..., :64]  (32 rows x 16 float4)
    {
        const float4* xs = (const float4*)x;
        float4* od = (float4*)out;
#pragma unroll
        for (int i = 0; i < 2; ++i) {
            int idx = tid + i * 256;           // 0..511
            int r = idx >> 4, q = idx & 15;    // 16 float4 per 64-float half-row
            od[(size_t)(rowBase + r) * 32 + q] = xs[(size_t)(rowBase + r) * 32 + q];
        }
    }

    // GEMM1: h(32x256) = x0(32x64) * w1^T(64x256); waves 0-3 strip 0, 4-7 strip 1
    {
        const int s = wave >> 2, ct0 = wave & 3;
        v16h a0 = load_a_f32cvt(x, DIM, rowBase + s * 16, 0,  lane);
        v16h a1 = load_a_f32cvt(x, DIM, rowBase + s * 16, 32, lane);
#pragma unroll
        for (int ct = ct0; ct < 16; ct += 4) {
            // loads first, then both WMMAs back-to-back
            v16h b0 = load_b_f16(w1h, HALFD, ct * 16, 0, lane);
            v16h b1 = load_b_f16(w1h, HALFD, ct * 16, 32, lane);
            v8f c = {};
            c = __builtin_amdgcn_wmma_f32_16x16x32_f16(false, a0, false, b0, (short)0, c, false, false);
            c = __builtin_amdgcn_wmma_f32_16x16x32_f16(false, a1, false, b1, (short)0, c, false, false);
            const int h = lane >> 4, n = lane & 15;
#pragma unroll
            for (int r = 0; r < 8; ++r)
                hbuf[(s * 16 + r + 8 * h) * HIDDEN + ct * 16 + n] = c[r];
        }
    }
    __syncthreads();

    // LayerNorm + ReLU + f16; each wave owns 4 rows, lane owns 8 contiguous cols
    for (int rr = 0; rr < 4; ++rr) {
        const int r = wave * 4 + rr;
        const float* hrow = &hbuf[r * HIDDEN];
        float s0 = 0.f, s1 = 0.f;
#pragma unroll
        for (int i = 0; i < 8; ++i) {
            float v = hrow[lane * 8 + i];
            s0 += v; s1 += v * v;
        }
#pragma unroll
        for (int off = 16; off > 0; off >>= 1) {
            s0 += __shfl_xor(s0, off, 32);
            s1 += __shfl_xor(s1, off, 32);
        }
        const float mean = s0 * (1.0f / HIDDEN);
        const float var  = s1 * (1.0f / HIDDEN) - mean * mean;
        const float rstd = rsqrtf(var + 1e-5f);
        alignas(16) _Float16 o[8];
#pragma unroll
        for (int i = 0; i < 8; ++i) {
            const int cc = lane * 8 + i;
            float v = (hrow[cc] - mean) * rstd * ln_scale[cc] + ln_bias[cc];
            o[i] = (_Float16)fmaxf(v, 0.f);
        }
        *(v8h*)(&hb[(size_t)(rowBase + r) * HIDDEN + lane * 8]) = *(const v8h*)o;
    }
}

// ---- kernel 2: GEMM2 (WMMA) fused with RQ-spline; stats never touch HBM ----
__global__ __launch_bounds__(256) void k_gemm2_spline(
        const float* __restrict__ x, const _Float16* __restrict__ hb,
        const _Float16* __restrict__ w2h, const float* __restrict__ b2,
        float* __restrict__ out, float* __restrict__ logdet) {
    extern __shared__ float sbuf[]; // 32 x 1472 f32 = 188416 B (fits 320KB WGP LDS)
    const int tid = threadIdx.x, lane = tid & 31, wave = tid >> 5;
    const int rowBase = blockIdx.x * RPB;

    // stats(32x1472) = hb(32x256) * w2^T(256x1472)
    {
        const int s = wave >> 2, ct0 = wave & 3;
        v16h a[8];
#pragma unroll
        for (int kc = 0; kc < 8; ++kc)
            a[kc] = load_a_f16(hb, HIDDEN, rowBase + s * 16, kc * 32, lane);

        const int h = lane >> 4, n = lane & 15;
        for (int ct = ct0; ct < 92; ct += 4) {  // 23 column tiles per wave
            // prefetch next tile's B slice for this lane (global_prefetch_b8)
            if (ct + 4 < 92)
                __builtin_prefetch(w2h + (size_t)((ct + 4) * 16 + n) * HIDDEN + h * 16, 0, 1);

            // stage the full 256x16 B tile into registers first ...
            v16h b[8];
#pragma unroll
            for (int kc = 0; kc < 8; ++kc)
                b[kc] = load_b_f16(w2h, HIDDEN, ct * 16, kc * 32, lane);

            // ... then 8 WMMAs back-to-back on two independent chains
            v8f c0 = {}, c1 = {};
#pragma unroll
            for (int kc = 0; kc < 8; kc += 2) {
                c0 = __builtin_amdgcn_wmma_f32_16x16x32_f16(false, a[kc],     false, b[kc],     (short)0, c0, false, false);
                c1 = __builtin_amdgcn_wmma_f32_16x16x32_f16(false, a[kc + 1], false, b[kc + 1], (short)0, c1, false, false);
            }
#pragma unroll
            for (int r = 0; r < 8; ++r)
                sbuf[(s * 16 + r + 8 * h) * OUTW + ct * 16 + n] = c0[r] + c1[r];
        }
    }
    __syncthreads();

    // spline: thread owns channel c = tid&63, rows rq*8 .. rq*8+7
    const int c = tid & 63, rq = tid >> 6;
    float bb[NSTAT];
#pragma unroll
    for (int j = 0; j < NSTAT; ++j) bb[j] = b2[c * NSTAT + j];

    for (int i = 0; i < 8; ++i) {
        const int r = rq * 8 + i;
        const float* sp = &sbuf[r * OUTW + c * NSTAT];
        float st[NSTAT];
#pragma unroll
        for (int j = 0; j < NSTAT; ++j) st[j] = sp[j] + bb[j];

        const float inv_sq = 1.0f / 16.0f; // 1/sqrt(HIDDEN)
        // widths: softmax(st[0..7]/16) -> knots cw[0..8]
        float cw[9], wid[8];
        {
            float mx = st[0] * inv_sq;
#pragma unroll
            for (int j = 1; j < 8; ++j) mx = fmaxf(mx, st[j] * inv_sq);
            float e[8], sum = 0.f;
#pragma unroll
            for (int j = 0; j < 8; ++j) { e[j] = __expf(st[j] * inv_sq - mx); sum += e[j]; }
            const float inv = 1.f / sum;
            float run = 0.f;
            cw[0] = -4.f;
#pragma unroll
            for (int j = 0; j < 8; ++j) {
                run += 0.001f + 0.992f * e[j] * inv;
                cw[j + 1] = 8.f * run - 4.f;
            }
            cw[8] = 4.f;
#pragma unroll
            for (int j = 0; j < 8; ++j) wid[j] = cw[j + 1] - cw[j];
        }
        // heights: same on st[8..15]
        float chh[9], hgt[8];
        {
            float mx = st[8] * inv_sq;
#pragma unroll
            for (int j = 1; j < 8; ++j) mx = fmaxf(mx, st[8 + j] * inv_sq);
            float e[8], sum = 0.f;
#pragma unroll
            for (int j = 0; j < 8; ++j) { e[j] = __expf(st[8 + j] * inv_sq - mx); sum += e[j]; }
            const float inv = 1.f / sum;
            float run = 0.f;
            chh[0] = -4.f;
#pragma unroll
            for (int j = 0; j < 8; ++j) {
                run += 0.001f + 0.992f * e[j] * inv;
                chh[j + 1] = 8.f * run - 4.f;
            }
            chh[8] = 4.f;
#pragma unroll
            for (int j = 0; j < 8; ++j) hgt[j] = chh[j + 1] - chh[j];
        }
        // derivatives: edges are exactly 1.0; inner = MIN_D + softplus(st[16+j])
        float dv[9];
        dv[0] = 1.0f; dv[8] = 1.0f;
#pragma unroll
        for (int j = 0; j < 7; ++j) {
            const float u = st[16 + j];
            const float spv = (u > 0.f) ? (u + log1pf(__expf(-u))) : log1pf(__expf(u));
            dv[j + 1] = 0.001f + spv;
        }

        const float xin = x[(size_t)(rowBase + r) * DIM + HALFD + c];
        // bin = clip(sum(xin >= locs) - 1, 0, 7), locs = cw with last += 1e-6
        int cnt = 0;
#pragma unroll
        for (int j = 0; j < 8; ++j) cnt += (xin >= cw[j]) ? 1 : 0;
        cnt += (xin >= cw[8] + 1e-6f) ? 1 : 0;
        int bin = cnt - 1;
        bin = bin < 0 ? 0 : (bin > 7 ? 7 : bin);

        // register-resident gathers (unrolled selects; no scratch spills)
        float icw = 0.f, iw = 0.f, ich = 0.f, ih = 0.f, id0 = 0.f, id1 = 0.f;
#pragma unroll
        for (int j = 0; j < 8; ++j) {
            const bool m = (bin == j);
            icw = m ? cw[j]     : icw;
            iw  = m ? wid[j]    : iw;
            ich = m ? chh[j]    : ich;
            ih  = m ? hgt[j]    : ih;
            id0 = m ? dv[j]     : id0;
            id1 = m ? dv[j + 1] : id1;
        }

        const float delta = ih / iw;
        const float theta = (xin - icw) / iw;
        const float t1mt  = theta * (1.f - theta);
        const float num   = ih * (delta * theta * theta + id0 * t1mt);
        const float den   = delta + (id0 + id1 - 2.f * delta) * t1mt;
        float y = ich + num / den;
        const float omt = 1.f - theta;
        const float dnum = delta * delta * (id1 * theta * theta + 2.f * delta * t1mt + id0 * omt * omt);
        float lad = __logf(dnum) - 2.f * __logf(den);

        const bool inside = (xin >= -4.f) && (xin <= 4.f);
        y   = inside ? y : xin;
        lad = inside ? lad : 0.f;

        out[(size_t)(rowBase + r) * DIM + HALFD + c] = y;
        logdet[(size_t)(rowBase + r) * HALFD + c] = lad;
    }
}

extern "C" void kernel_launch(void* const* d_in, const int* in_sizes, int n_in,
                              void* d_out, int out_size, void* d_ws, size_t ws_size,
                              hipStream_t stream) {
    const float* x   = (const float*)d_in[0];
    const float* w1  = (const float*)d_in[1];
    const float* lns = (const float*)d_in[2];
    const float* lnb = (const float*)d_in[3];
    const float* w2  = (const float*)d_in[4];
    const float* b2  = (const float*)d_in[5];

    float* out    = (float*)d_out;
    float* logdet = out + (size_t)ROWS_TOTAL * DIM;

    // workspace layout (~17.6 MB total):
    //   [0, 32KB)        w1h  f16  256x64
    //   [32KB, 768KB)    w2h  f16  1472x256
    //   [768KB, +16MB)   hb   f16  32768x256
    _Float16* w1h = (_Float16*)d_ws;
    _Float16* w2h = w1h + HIDDEN * HALFD;
    _Float16* hb  = (_Float16*)((char*)d_ws + 786432);

    k_cvt<<<(OUTW * HIDDEN + 255) / 256, 256, 0, stream>>>(w1, w2, w1h, w2h);
    k_gemm1_ln<<<ROWS_TOTAL / RPB, 256, 0, stream>>>(x, w1h, lns, lnb, hb, out);
    k_gemm2_spline<<<ROWS_TOTAL / RPB, 256, RPB * OUTW * sizeof(float), stream>>>(
        x, hb, w2h, b2, out, logdet);
}
